// FFNO_SMM_10591389352468
// MI455X (gfx1250) — matrix operand order repaired and checked by reading, not compiled, not run on hardware
//
#include <hip/hip_runtime.h>
#include <hip/hip_bf16.h>
#include <math.h>

typedef _Float16 f16;
typedef __attribute__((ext_vector_type(16))) _Float16 v16h;
typedef __attribute__((ext_vector_type(8)))  float    v8f;

#define BB   16
#define SY   180
#define SX   360
#define CH   32
#define MD   32
#define SYP  192   // SY padded to multiple of 32
#define SXP  384   // SX padded to multiple of 32
#define NPIX (SY * SX)  // 64800 = 2025 * 32

// Padded leading dims for transposed-B LDS tiles (keep 16B alignment: ld*2 % 16 == 0)
#define LDB32  40    // K=32 tiles
#define LDB128 144   // K=128 tiles
#define LDB192 208   // K=192 tiles

#define WMMA(a, b, c) __builtin_amdgcn_wmma_f32_16x16x32_f16( \
    false, (a), false, (b), (short)0, (c), false, false)

// ---------------------------------------------------------------------------
// Fragment builders (CDNA5 wave32 16x16x32 f16 layouts), operands staged in LDS.
// A: 16 x K row-major (row = lane&15), element e -> K = (e>>3)*16 + hi*8 + (e&7)
//    -> two contiguous 8-half runs per lane => 2x ds_load_b128.
// B: staged TRANSPOSED in LDS as [col][k]; element e -> K = (e>>3)*16 + hi*8 + (e&7)
//    -> two contiguous 8-half runs per lane => 2x ds_load_b128.
// D: f32 16x16, VGPR v: m = v + 8*hi, n = lane&15
// ---------------------------------------------------------------------------
__device__ __forceinline__ v16h frag_A(const f16* p, int ld) {
  const int ln = threadIdx.x & 31, r = ln & 15, hi = ln >> 4;
  v16h a;
#pragma unroll
  for (int e = 0; e < 16; ++e) {
    const int k = ((e >> 3) << 4) + (hi << 3) + (e & 7);
    a[e] = p[r * ld + k];
  }
  return a;
}

// p points at transposed tile base: element (k, col) lives at p[col*ld + k]
__device__ __forceinline__ v16h frag_BT(const f16* p, int ld) {
  const int ln = threadIdx.x & 31, c = ln & 15, hi = ln >> 4;
  const f16* q = p + c * ld + (hi << 3);
  v16h b;
#pragma unroll
  for (int e = 0; e < 16; ++e) b[e] = q[((e >> 3) << 4) + (e & 7)];
  return b;
}

__device__ __forceinline__ float gelu_f(float v) {
  return 0.5f * v * (1.0f + erff(v * 0.7071067811865475f));
}

// ---------------------------------------------------------------------------
// Prep: grid coords + Vandermonde matrices (f16, padded) and their transposes
// ---------------------------------------------------------------------------
__global__ void k_prep(const float* __restrict__ sy, const float* __restrict__ sx,
                       float* __restrict__ gy, float* __restrict__ gx,
                       f16* VyRe, f16* VyIm, f16* VyTRe, f16* VyTIm,
                       f16* VxRe, f16* VxIm, f16* VxTRe, f16* VxTIm) {
  const int t = threadIdx.x;
  const float y0 = sy[0], yN = sy[SY - 1];
  const float x0 = sx[0], xN = sx[SX - 1];
  for (int i = t; i < SY; i += 256) gy[i] = (sy[i] - y0) / yN;
  for (int i = t; i < SX; i += 256) gx[i] = (sx[i] - x0) / xN;
  const float TWO_PI = 6.2831853071795864769f;
  const float scy = rsqrtf((float)SY), scx = rsqrtf((float)SX);
  for (int idx = t; idx < MD * SYP; idx += 256) {
    const int k = idx / SYP, m = idx % SYP;
    float re = 0.f, im = 0.f;
    if (m < SY) {
      const float p = (sy[m] - y0) / (yN - y0);
      const float ang = TWO_PI * (float)k * p;
      re = cosf(ang) * scy;
      im = -sinf(ang) * scy;
    }
    VyRe[idx] = (f16)re; VyIm[idx] = (f16)im;
    VyTRe[m * MD + k] = (f16)re; VyTIm[m * MD + k] = (f16)im;
  }
  for (int idx = t; idx < MD * SXP; idx += 256) {
    const int k = idx / SXP, n = idx % SXP;
    float re = 0.f, im = 0.f;
    if (n < SX) {
      const float p = (sx[n] - x0) / (xN - x0);
      const float ang = TWO_PI * (float)k * p;
      re = cosf(ang) * scx;
      im = -sinf(ang) * scx;
    }
    VxRe[idx] = (f16)re; VxIm[idx] = (f16)im;
    VxTRe[n * MD + k] = (f16)re; VxTIm[n * MD + k] = (f16)im;
  }
}

// ---------------------------------------------------------------------------
// fc0: [x | gx | gy] @ fc0_w + b -> h32 [B,32,SY,SX] (+ f16 mirror)
// ---------------------------------------------------------------------------
__global__ void k_fc0(const float* __restrict__ x, const float* __restrict__ gx,
                      const float* __restrict__ gy, const float* __restrict__ w,
                      const float* __restrict__ bias,
                      float* __restrict__ h32, f16* __restrict__ h16) {
  __shared__ float sw[18 * 32];
  __shared__ float sb[32];
  const int t = threadIdx.x;
  for (int i = t; i < 18 * 32; i += 256) sw[i] = w[i];
  if (t < 32) sb[t] = bias[t];
  __syncthreads();
  const long p = (long)blockIdx.x * 256 + t;
  if (p >= (long)BB * NPIX) return;
  const int b = (int)(p / NPIX), r = (int)(p % NPIX);
  const int yy = r / SX, xx = r % SX;
  float in[18];
  const float* xp = x + p * 16;
#pragma unroll
  for (int i = 0; i < 16; ++i) in[i] = xp[i];
  in[16] = gx[xx];
  in[17] = gy[yy];
#pragma unroll 4
  for (int o = 0; o < 32; ++o) {
    float acc = sb[o];
#pragma unroll
    for (int i = 0; i < 18; ++i) acc += in[i] * sw[i * 32 + o];
    const long addr = (long)(b * 32 + o) * NPIX + r;
    h32[addr] = acc;
    h16[addr] = (f16)acc;
  }
}

// ---------------------------------------------------------------------------
// Forward-y: per (b,c):  fty[k,n] = sum_m Vy[k,m] * h[m,n]   (Re & Im)
// grid (12 ntiles, B*CH), 128 threads. Output fty* [B,C,MD,SX] f16.
// ---------------------------------------------------------------------------
__global__ void k_fwdy(const f16* __restrict__ h16, const f16* __restrict__ VyRe,
                       const f16* __restrict__ VyIm,
                       f16* __restrict__ ftyRe, f16* __restrict__ ftyIm) {
  __shared__ f16 sA[2 * MD * SYP];      // VyRe | VyIm (row-major [k][m])
  __shared__ f16 sBT[32 * LDB192];      // h slab transposed: [n_local][m]
  const int t = threadIdx.x;
  const int n0 = blockIdx.x * 32;
  const int bc = blockIdx.y;
  for (int i = t; i < 2 * MD * SYP; i += 128)
    sA[i] = (i < MD * SYP) ? VyRe[i] : VyIm[i - MD * SYP];
  const long hbase = (long)bc * NPIX;
  for (int i = t; i < SYP * 32; i += 128) {
    const int m = i >> 5, nl = i & 31;
    const int n = n0 + nl;
    sBT[nl * LDB192 + m] =
        (m < SY && n < SX) ? h16[hbase + (long)m * SX + n] : (f16)0.f;
  }
  __syncthreads();
  const int w = t >> 5, ln = t & 31, hi = ln >> 4, nn = ln & 15;
  const int rt = w >> 1, ct = w & 1;
  v8f accR = {0, 0, 0, 0, 0, 0, 0, 0};
  v8f accI = {0, 0, 0, 0, 0, 0, 0, 0};
#pragma unroll
  for (int kk = 0; kk < SYP; kk += 32) {
    const v16h aR = frag_A(sA + rt * 16 * SYP + kk, SYP);
    const v16h aI = frag_A(sA + MD * SYP + rt * 16 * SYP + kk, SYP);
    const v16h bb = frag_BT(sBT + ct * 16 * LDB192 + kk, LDB192);
    accR = WMMA(aR, bb, accR);
    accI = WMMA(aI, bb, accI);
  }
  const int n = n0 + ct * 16 + nn;
  if (n < SX) {
#pragma unroll
    for (int v = 0; v < 8; ++v) {
      const int k = rt * 16 + v + 8 * hi;
      const long o = (long)(bc * MD + k) * SX + n;
      ftyRe[o] = (f16)accR[v];
      ftyIm[o] = (f16)accI[v];
    }
  }
}

// ---------------------------------------------------------------------------
// Complex per-mode channel mix (shared by y and x directions).
// per (b,k): out[o,col] = sum_i w[i,o,k] * src[i,col]  (complex)
// B element addr = b*32*iStride + k*kStride + i*iStride + col*colStride
// out addr       = ((b*32+k)*32 + o)*L + col
// ---------------------------------------------------------------------------
__global__ void k_mix(const f16* __restrict__ bRe, const f16* __restrict__ bIm,
                      const float* __restrict__ w4,
                      f16* __restrict__ oRe, f16* __restrict__ oIm,
                      int L, int iStride, int kStride, int colStride) {
  __shared__ f16 sAr[1024], sAi[1024];
  __shared__ f16 sBrT[32 * LDB32], sBiT[32 * LDB32];  // [col][i]
  const int t = threadIdx.x;
  const int c0 = blockIdx.x * 32;
  const int b = blockIdx.y >> 5, k = blockIdx.y & 31;
  for (int idx = t; idx < 1024; idx += 128) {
    const int o = idx >> 5, i = idx & 31;
    const float* s = w4 + ((long)(i * 32 + o) * MD + k) * 2;
    sAr[idx] = (f16)s[0];
    sAi[idx] = (f16)s[1];
  }
  const long base0 = (long)(b * 32) * iStride + (long)k * kStride;
  for (int idx = t; idx < 1024; idx += 128) {
    const int i = idx >> 5, col = idx & 31;
    const int gc = c0 + col;
    f16 vr = (f16)0.f, vi = (f16)0.f;
    if (gc < L) {
      const long a = base0 + (long)i * iStride + (long)gc * colStride;
      vr = bRe[a];
      vi = bIm[a];
    }
    sBrT[col * LDB32 + i] = vr;
    sBiT[col * LDB32 + i] = vi;
  }
  __syncthreads();
  const int w = t >> 5, ln = t & 31, hi = ln >> 4, nn = ln & 15;
  const int rt = w >> 1, ct = w & 1;
  const v16h aR = frag_A(sAr + rt * 16 * 32, 32);
  const v16h aI = frag_A(sAi + rt * 16 * 32, 32);
  v16h aIn;
#pragma unroll
  for (int e = 0; e < 16; ++e) aIn[e] = (f16)(-(float)aI[e]);
  const v16h bR = frag_BT(sBrT + ct * 16 * LDB32, LDB32);
  const v16h bI = frag_BT(sBiT + ct * 16 * LDB32, LDB32);
  const v8f z = {0, 0, 0, 0, 0, 0, 0, 0};
  const v8f accR = WMMA(aR, bR, WMMA(aIn, bI, z));   // Re*Re - Im*Im
  const v8f accI = WMMA(aR, bI, WMMA(aI, bR, z));    // Re*Im + Im*Re
  const int col = c0 + ct * 16 + nn;
  if (col < L) {
#pragma unroll
    for (int v = 0; v < 8; ++v) {
      const int o = rt * 16 + v + 8 * hi;
      const long oa = ((long)(b * 32 + k) * 32 + o) * L + col;
      oRe[oa] = (f16)accR[v];
      oIm[oa] = (f16)accI[v];
    }
  }
}

// ---------------------------------------------------------------------------
// Inverse-y: per (b,o): x1[m,n] = sum_k VyT[m,k]Re*oyRe[k,n] + Im*Im  (write)
// grid (12 ntiles, 6 mtiles, B*CH)
// ---------------------------------------------------------------------------
__global__ void k_invy(const f16* __restrict__ VyTRe, const f16* __restrict__ VyTIm,
                       const f16* __restrict__ oyRe, const f16* __restrict__ oyIm,
                       float* __restrict__ x1) {
  __shared__ f16 sAr[1024], sAi[1024];
  __shared__ f16 sBrT[32 * LDB32], sBiT[32 * LDB32];  // [n_local][k]
  const int t = threadIdx.x;
  const int n0 = blockIdx.x * 32, m0 = blockIdx.y * 32;
  const int b = blockIdx.z >> 5, o = blockIdx.z & 31;
  for (int idx = t; idx < 1024; idx += 128) {
    const int mr = idx >> 5, k = idx & 31;
    sAr[idx] = VyTRe[(m0 + mr) * MD + k];
    sAi[idx] = VyTIm[(m0 + mr) * MD + k];
  }
  for (int idx = t; idx < 1024; idx += 128) {
    const int k = idx >> 5, c = idx & 31;
    const int gn = n0 + c;
    f16 vr = (f16)0.f, vi = (f16)0.f;
    if (gn < SX) {
      const long a = ((long)(b * 32 + k) * 32 + o) * SX + gn;
      vr = oyRe[a];
      vi = oyIm[a];
    }
    sBrT[c * LDB32 + k] = vr;
    sBiT[c * LDB32 + k] = vi;
  }
  __syncthreads();
  const int w = t >> 5, ln = t & 31, hi = ln >> 4, nn = ln & 15;
  const int rt = w >> 1, ct = w & 1;
  const v16h aR = frag_A(sAr + rt * 16 * 32, 32);
  const v16h aI = frag_A(sAi + rt * 16 * 32, 32);
  const v16h bR = frag_BT(sBrT + ct * 16 * LDB32, LDB32);
  const v16h bI = frag_BT(sBiT + ct * 16 * LDB32, LDB32);
  const v8f z = {0, 0, 0, 0, 0, 0, 0, 0};
  const v8f acc = WMMA(aI, bI, WMMA(aR, bR, z));
  const int gn = n0 + ct * 16 + nn;
#pragma unroll
  for (int v = 0; v < 8; ++v) {
    const int gm = m0 + rt * 16 + v + 8 * hi;
    if (gm < SY && gn < SX)
      x1[(long)(b * 32 + o) * NPIX + (long)gm * SX + gn] = acc[v];
  }
}

// ---------------------------------------------------------------------------
// Forward-x: per (b,c): ftx[m,k] = sum_n h[m,n] * VxT[n,k]  (Re & Im)
// grid (6 mtiles, B*CH). Contraction over n chunked through LDS (128 at a time).
// Output ftx* [B,C,SY,MD] f16.
// ---------------------------------------------------------------------------
__global__ void k_fwdx(const f16* __restrict__ h16, const f16* __restrict__ VxTRe,
                       const f16* __restrict__ VxTIm,
                       f16* __restrict__ ftxRe, f16* __restrict__ ftxIm) {
  __shared__ f16 sA[32 * 128];                           // [m_local][n_chunk]
  __shared__ f16 sBrT[32 * LDB128], sBiT[32 * LDB128];   // [k][n_chunk]
  const int t = threadIdx.x;
  const int m0 = blockIdx.x * 32;
  const int bc = blockIdx.y;
  const int w = t >> 5, ln = t & 31, hi = ln >> 4, nn = ln & 15;
  const int rt = w >> 1, ct = w & 1;
  v8f accR = {0, 0, 0, 0, 0, 0, 0, 0};
  v8f accI = {0, 0, 0, 0, 0, 0, 0, 0};
  const long hbase = (long)bc * NPIX;
  for (int nc = 0; nc < SXP; nc += 128) {
    __syncthreads();
    for (int idx = t; idx < 32 * 128; idx += 128) {
      const int mr = idx >> 7, n = nc + (idx & 127);
      const int gm = m0 + mr;
      sA[idx] = (gm < SY && n < SX) ? h16[hbase + (long)gm * SX + n] : (f16)0.f;
    }
    for (int idx = t; idx < 128 * 32; idx += 128) {
      const int nr = idx >> 5, k = idx & 31;
      const int n = nc + nr;
      sBrT[k * LDB128 + nr] = VxTRe[n * MD + k];
      sBiT[k * LDB128 + nr] = VxTIm[n * MD + k];
    }
    __syncthreads();
#pragma unroll
    for (int kk = 0; kk < 128; kk += 32) {
      const v16h a  = frag_A(sA + rt * 16 * 128 + kk, 128);
      const v16h bR = frag_BT(sBrT + ct * 16 * LDB128 + kk, LDB128);
      const v16h bI = frag_BT(sBiT + ct * 16 * LDB128 + kk, LDB128);
      accR = WMMA(a, bR, accR);
      accI = WMMA(a, bI, accI);
    }
  }
  const int kmode = ct * 16 + nn;
#pragma unroll
  for (int v = 0; v < 8; ++v) {
    const int gm = m0 + rt * 16 + v + 8 * hi;
    if (gm < SY) {
      const long a = ((long)bc * SY + gm) * MD + kmode;
      ftxRe[a] = (f16)accR[v];
      ftxIm[a] = (f16)accI[v];
    }
  }
}

// ---------------------------------------------------------------------------
// Inverse-x: per (b,o): x1[m,n] += sum_k oxRe[k,m]*VxRe[k,n] + Im*Im
// grid (12 ntiles, 6 mtiles, B*CH)
// ---------------------------------------------------------------------------
__global__ void k_invx(const f16* __restrict__ oxRe, const f16* __restrict__ oxIm,
                       const f16* __restrict__ VxRe, const f16* __restrict__ VxIm,
                       float* __restrict__ x1) {
  __shared__ f16 sAr[1024], sAi[1024];
  __shared__ f16 sBrT[32 * LDB32], sBiT[32 * LDB32];  // [n_local][k]
  const int t = threadIdx.x;
  const int n0 = blockIdx.x * 32, m0 = blockIdx.y * 32;
  const int b = blockIdx.z >> 5, o = blockIdx.z & 31;
  for (int idx = t; idx < 1024; idx += 128) {
    const int mr = idx >> 5, k = idx & 31;
    const int gm = m0 + mr;
    f16 vr = (f16)0.f, vi = (f16)0.f;
    if (gm < SY) {
      const long a = ((long)(b * 32 + k) * 32 + o) * SY + gm;
      vr = oxRe[a];
      vi = oxIm[a];
    }
    sAr[idx] = vr;
    sAi[idx] = vi;
  }
  for (int idx = t; idx < 1024; idx += 128) {
    const int k = idx >> 5, c = idx & 31;
    const int gn = n0 + c;  // < 384 always; Vx buffers zero-padded
    sBrT[c * LDB32 + k] = VxRe[k * SXP + gn];
    sBiT[c * LDB32 + k] = VxIm[k * SXP + gn];
  }
  __syncthreads();
  const int w = t >> 5, ln = t & 31, hi = ln >> 4, nn = ln & 15;
  const int rt = w >> 1, ct = w & 1;
  const v16h aR = frag_A(sAr + rt * 16 * 32, 32);
  const v16h aI = frag_A(sAi + rt * 16 * 32, 32);
  const v16h bR = frag_BT(sBrT + ct * 16 * LDB32, LDB32);
  const v16h bI = frag_BT(sBiT + ct * 16 * LDB32, LDB32);
  const v8f z = {0, 0, 0, 0, 0, 0, 0, 0};
  const v8f acc = WMMA(aI, bI, WMMA(aR, bR, z));
  const int gn = n0 + ct * 16 + nn;
#pragma unroll
  for (int v = 0; v < 8; ++v) {
    const int gm = m0 + rt * 16 + v + 8 * hi;
    if (gm < SY && gn < SX) {
      const long a = (long)(b * 32 + o) * NPIX + (long)gm * SX + gn;
      x1[a] += acc[v];
    }
  }
}

// ---------------------------------------------------------------------------
// Pointwise block: x2 = W1*x1+b1; t = gelu(x2); x4 = W2*t+b2;
// h = (gelu(x4) if !last else x4) + h ;  rewrite h32 + h16 mirror.
// grid (NPIX/32, B), 128 threads.
// ---------------------------------------------------------------------------
__global__ void k_pw(const float* __restrict__ x1, const float* __restrict__ w1,
                     const float* __restrict__ b1, const float* __restrict__ w2,
                     const float* __restrict__ b2,
                     float* __restrict__ h32, f16* __restrict__ h16, int lastBlock) {
  __shared__ f16 sA1[1024], sA2[1024];
  __shared__ f16 sBT[32 * LDB32];   // x1 tile transposed: [p][c]
  __shared__ f16 sTT[32 * LDB32];   // gelu(x2) tile transposed: [p][o]
  __shared__ float sb1[32], sb2[32];
  const int t = threadIdx.x;
  const int p0 = blockIdx.x * 32;
  const int b = blockIdx.y;
  for (int idx = t; idx < 1024; idx += 128) {
    sA1[idx] = (f16)w1[idx];  // [o,c] row-major, matches A layout directly
    sA2[idx] = (f16)w2[idx];
  }
  if (t < 32) { sb1[t] = b1[t]; sb2[t] = b2[t]; }
  for (int idx = t; idx < 1024; idx += 128) {
    const int c = idx >> 5, pp = idx & 31;
    sBT[pp * LDB32 + c] = (f16)x1[(long)(b * 32 + c) * NPIX + p0 + pp];
  }
  __syncthreads();
  const int w = t >> 5, ln = t & 31, hi = ln >> 4, nn = ln & 15;
  const int rt = w >> 1, ct = w & 1;
  const v8f z = {0, 0, 0, 0, 0, 0, 0, 0};
  {
    const v16h a = frag_A(sA1 + rt * 16 * 32, 32);
    const v16h bb = frag_BT(sBT + ct * 16 * LDB32, LDB32);
    const v8f acc = WMMA(a, bb, z);
    const int pc = ct * 16 + nn;
#pragma unroll
    for (int v = 0; v < 8; ++v) {
      const int o = rt * 16 + v + 8 * hi;
      sTT[pc * LDB32 + o] = (f16)gelu_f(acc[v] + sb1[o]);
    }
  }
  __syncthreads();
  {
    const v16h a = frag_A(sA2 + rt * 16 * 32, 32);
    const v16h bb = frag_BT(sTT + ct * 16 * LDB32, LDB32);
    const v8f acc = WMMA(a, bb, z);
    const int pc = ct * 16 + nn;
    const int gp = p0 + pc;
#pragma unroll
    for (int v = 0; v < 8; ++v) {
      const int o = rt * 16 + v + 8 * hi;
      float val = acc[v] + sb2[o];
      if (!lastBlock) val = gelu_f(val);
      const long addr = (long)(b * 32 + o) * NPIX + gp;
      const float nh = val + h32[addr];
      h32[addr] = nh;
      h16[addr] = (f16)nh;
    }
  }
}

// ---------------------------------------------------------------------------
// Final head: hidden = gelu(fc1^T * h + b1) [128 x 32p]; out = fc2 . hidden + b2
// grid (NPIX/32, B), 256 threads (8 waves: one 16-row tile each, 2 col tiles).
// ---------------------------------------------------------------------------
__global__ void k_final(const float* __restrict__ h32, const float* __restrict__ fc1w,
                        const float* __restrict__ fc1b, const float* __restrict__ fc2w,
                        const float* __restrict__ fc2b, float* __restrict__ out) {
  __shared__ f16 sA[128 * 32];
  __shared__ f16 sBT[32 * LDB32];   // h tile transposed: [p][c]
  __shared__ float sHid[128 * 32], sb1[128], sw2[128];
  const int t = threadIdx.x;
  const int p0 = blockIdx.x * 32;
  const int b = blockIdx.y;
  for (int idx = t; idx < 128 * 32; idx += 256) {
    const int h = idx >> 5, c = idx & 31;
    sA[idx] = (f16)fc1w[c * 128 + h];  // transpose: A[h,c]
  }
  for (int idx = t; idx < 1024; idx += 256) {
    const int c = idx >> 5, pp = idx & 31;
    sBT[pp * LDB32 + c] = (f16)h32[(long)(b * 32 + c) * NPIX + p0 + pp];
  }
  if (t < 128) { sb1[t] = fc1b[t]; sw2[t] = fc2w[t]; }
  __syncthreads();
  const int w = t >> 5, ln = t & 31, hi = ln >> 4, nn = ln & 15;
  const v16h a = frag_A(sA + w * 16 * 32, 32);
  const v8f z = {0, 0, 0, 0, 0, 0, 0, 0};
  const v8f acc0 = WMMA(a, frag_BT(sBT, LDB32), z);
  const v8f acc1 = WMMA(a, frag_BT(sBT + 16 * LDB32, LDB32), z);
#pragma unroll
  for (int v = 0; v < 8; ++v) {
    const int h = w * 16 + v + 8 * hi;
    sHid[h * 32 + nn]      = gelu_f(acc0[v] + sb1[h]);
    sHid[h * 32 + 16 + nn] = gelu_f(acc1[v] + sb1[h]);
  }
  __syncthreads();
  if (t < 32) {
    float acc = fc2b[0];
#pragma unroll 8
    for (int h = 0; h < 128; ++h) acc += sHid[h * 32 + t] * sw2[h];
    out[(long)b * NPIX + p0 + t] = acc;
  }
}

// ---------------------------------------------------------------------------
// Host launcher
// ---------------------------------------------------------------------------
extern "C" void kernel_launch(void* const* d_in, const int* in_sizes, int n_in,
                              void* d_out, int out_size, void* d_ws, size_t ws_size,
                              hipStream_t stream) {
  (void)in_sizes; (void)n_in; (void)out_size;
  const float* x    = (const float*)d_in[0];
  const float* spx  = (const float*)d_in[1];
  const float* spy  = (const float*)d_in[2];
  const float* fc0w = (const float*)d_in[3];
  const float* fc0b = (const float*)d_in[4];
  const float* fw1  = (const float*)d_in[5];
  const float* fw2  = (const float*)d_in[6];
  const float* w1   = (const float*)d_in[7];
  const float* b1   = (const float*)d_in[8];
  const float* w2   = (const float*)d_in[9];
  const float* b2   = (const float*)d_in[10];
  const float* fc1w = (const float*)d_in[11];
  const float* fc1b = (const float*)d_in[12];
  const float* fc2w = (const float*)d_in[13];
  const float* fc2b = (const float*)d_in[14];
  float* out = (float*)d_out;

  char* ws = (char*)d_ws;
  size_t off = 0;
  auto take = [&](size_t bytes) -> char* {
    char* p = ws + off;
    off = (off + bytes + 255) & ~(size_t)255;
    return p;
  };
  float* gy    = (float*)take((size_t)SY * 4);
  float* gx    = (float*)take((size_t)SX * 4);
  f16* VyRe    = (f16*)take((size_t)MD * SYP * 2);
  f16* VyIm    = (f16*)take((size_t)MD * SYP * 2);
  f16* VyTRe   = (f16*)take((size_t)SYP * MD * 2);
  f16* VyTIm   = (f16*)take((size_t)SYP * MD * 2);
  f16* VxRe    = (f16*)take((size_t)MD * SXP * 2);
  f16* VxIm    = (f16*)take((size_t)MD * SXP * 2);
  f16* VxTRe   = (f16*)take((size_t)SXP * MD * 2);
  f16* VxTIm   = (f16*)take((size_t)SXP * MD * 2);
  float* h32   = (float*)take((size_t)BB * CH * NPIX * 4);
  f16* h16     = (f16*)take((size_t)BB * CH * NPIX * 2);
  float* x1    = (float*)take((size_t)BB * CH * NPIX * 4);
  f16* ftyRe   = (f16*)take((size_t)BB * CH * MD * SX * 2);
  f16* ftyIm   = (f16*)take((size_t)BB * CH * MD * SX * 2);
  f16* oyRe    = (f16*)take((size_t)BB * MD * CH * SX * 2);
  f16* oyIm    = (f16*)take((size_t)BB * MD * CH * SX * 2);
  f16* ftxRe   = (f16*)take((size_t)BB * CH * SY * MD * 2);
  f16* ftxIm   = (f16*)take((size_t)BB * CH * SY * MD * 2);
  f16* oxRe    = (f16*)take((size_t)BB * MD * CH * SY * 2);
  f16* oxIm    = (f16*)take((size_t)BB * MD * CH * SY * 2);
  if (off > ws_size) return;  // workspace too small; cannot run safely

  k_prep<<<1, 256, 0, stream>>>(spy, spx, gy, gx, VyRe, VyIm, VyTRe, VyTIm,
                                VxRe, VxIm, VxTRe, VxTIm);
  const int npts = BB * NPIX;
  k_fc0<<<(npts + 255) / 256, 256, 0, stream>>>(x, gx, gy, fc0w, fc0b, h32, h16);

  for (int blk = 0; blk < 4; ++blk) {
    const float* fw1b = fw1 + (size_t)blk * CH * CH * MD * 2;
    const float* fw2b = fw2 + (size_t)blk * CH * CH * MD * 2;
    const float* w1b  = w1 + (size_t)blk * CH * CH;
    const float* w2b  = w2 + (size_t)blk * CH * CH;
    const float* b1b  = b1 + (size_t)blk * CH;
    const float* b2b  = b2 + (size_t)blk * CH;

    // y-direction
    k_fwdy<<<dim3(SXP / 32, BB * CH), 128, 0, stream>>>(h16, VyRe, VyIm, ftyRe, ftyIm);
    k_mix<<<dim3(SXP / 32, BB * MD), 128, 0, stream>>>(
        ftyRe, ftyIm, fw1b, oyRe, oyIm, SX, MD * SX, SX, 1);
    k_invy<<<dim3(SXP / 32, SYP / 32, BB * CH), 128, 0, stream>>>(
        VyTRe, VyTIm, oyRe, oyIm, x1);
    // x-direction
    k_fwdx<<<dim3(SYP / 32, BB * CH), 128, 0, stream>>>(h16, VxTRe, VxTIm, ftxRe, ftxIm);
    k_mix<<<dim3(SYP / 32, BB * MD), 128, 0, stream>>>(
        ftxRe, ftxIm, fw2b, oxRe, oxIm, SY, SY * MD, 1, MD);
    k_invx<<<dim3(SXP / 32, SYP / 32, BB * CH), 128, 0, stream>>>(
        oxRe, oxIm, VxRe, VxIm, x1);
    // pointwise + residual
    k_pw<<<dim3(NPIX / 32, BB), 128, 0, stream>>>(x1, w1b, b1b, w2b, b2b, h32, h16,
                                                  blk == 3 ? 1 : 0);
  }
  k_final<<<dim3(NPIX / 32, BB), 256, 0, stream>>>(h32, fc1w, fc1b, fc2w, fc2b, out);
}
